// BertAttention_76218489635381
// MI455X (gfx1250) — matrix operand order repaired and checked by reading, not compiled, not run on hardware
//
#include <hip/hip_runtime.h>
#include <stdint.h>

#define B_ 8
#define S_ 2048
#define H_ 768
#define LN_EPS 1e-12f

typedef __bf16 bf16;
typedef __attribute__((ext_vector_type(16))) __bf16 v16bf;
typedef __attribute__((ext_vector_type(8)))  __bf16 v8bf;
typedef __attribute__((ext_vector_type(8)))  float  v8f;

__device__ __forceinline__ bf16 to_bf16(float f) {
  uint32_t u = __builtin_bit_cast(uint32_t, f);
  u += 0x7FFFu + ((u >> 16) & 1u);          // round-to-nearest-even
  uint16_t hs = (uint16_t)(u >> 16);
  return __builtin_bit_cast(bf16, hs);
}
__device__ __forceinline__ float from_bf16(bf16 h) {
  uint16_t hs = __builtin_bit_cast(uint16_t, h);
  uint32_t u = ((uint32_t)hs) << 16;
  return __builtin_bit_cast(float, u);
}

// gfx1250 async global->LDS copy (ASYNCcnt-tracked, no VGPR data path).
// VDST = VGPR holding LDS byte address, VADDR = 64-bit global address.
__device__ __forceinline__ void async_copy_b128(uint32_t lds_off, const void* gaddr) {
  asm volatile("global_load_async_to_lds_b128 %0, %1, off"
               :: "v"(lds_off), "v"(gaddr) : "memory");
}
__device__ __forceinline__ void wait_async0() {
  asm volatile("s_wait_asynccnt 0x0" ::: "memory");
}

// A fragment: 16x32 bf16, lane = M (lanes 0-15 / 16-31 give K halves).
// elements 0..15 -> K = 16*half+0..15 (contiguous)
__device__ __forceinline__ v16bf load_a_frag(const bf16* p, int ld, int k0, int lane) {
  int half = lane >> 4, m = lane & 15;
  const bf16* r = p + (size_t)m * ld + k0 + 16 * half;
  v8bf lo = *(const v8bf*)(r);
  v8bf hi = *(const v8bf*)(r + 8);
  v16bf a;
#pragma unroll
  for (int i = 0; i < 8; ++i) { a[i] = lo[i]; a[i + 8] = hi[i]; }
  return a;
}

// B fragment: 32x16 bf16, lane = N, K contiguous per lane (0-15 -> K0..15, 16-31 -> K16..31).
// Loaded from "Bt" storage where row n holds B[:,n] contiguously along K.
__device__ __forceinline__ v16bf load_b_frag(const bf16* p, int ld, int k0, int lane) {
  int half = lane >> 4, n = lane & 15;
  return *(const v16bf*)(p + (size_t)n * ld + k0 + 16 * half);
}

__device__ __forceinline__ v8f wmma_bf16(v16bf a, v16bf b, v8f c) {
  return __builtin_amdgcn_wmma_f32_16x16x32_bf16(false, a, false, b, (short)0, c, false, false);
}

// ---------------- Kernel 1a: fp32 -> bf16 convert ----------------
__global__ void cvt_kernel(const float* __restrict__ X, bf16* __restrict__ Xb, int n) {
  int i = blockIdx.x * blockDim.x + threadIdx.x;
  int stride = gridDim.x * blockDim.x;
  for (; i < n; i += stride) Xb[i] = to_bf16(X[i]);
}

// ---------------- Kernel 1b: weight transpose + convert (W[in][out] -> Wt[out][in]) ----
__global__ void wtrans_kernel(const float* __restrict__ W, bf16* __restrict__ Wt) {
  __shared__ float t[16][17];
  int i0 = blockIdx.x * 16, o0 = blockIdx.y * 16;
  t[threadIdx.y][threadIdx.x] = W[(size_t)(i0 + threadIdx.y) * H_ + o0 + threadIdx.x];
  __syncthreads();
  Wt[(size_t)(o0 + threadIdx.y) * H_ + i0 + threadIdx.x] = to_bf16(t[threadIdx.x][threadIdx.y]);
}

// ---------------- Kernel 2: GEMM (X[BS,H] * W -> bf16 out, optionally V^T layout) ------
__global__ void __launch_bounds__(256)
gemm_bf16(const bf16* __restrict__ A, const bf16* __restrict__ Bt,
          const float* __restrict__ bias, bf16* __restrict__ OutRM,
          bf16* __restrict__ OutVt, float scale) {
  int lane = threadIdx.x & 31;
  int w = threadIdx.x >> 5;
  int m_base = blockIdx.x * 128 + (w & 3) * 32;
  int n_base = blockIdx.y * 64 + (w >> 2) * 32;

  v8f acc[2][2] = {};
  for (int k = 0; k < H_; k += 32) {
    v16bf a0 = load_a_frag(A + (size_t)m_base * H_, H_, k, lane);
    v16bf a1 = load_a_frag(A + (size_t)(m_base + 16) * H_, H_, k, lane);
    v16bf b0 = load_b_frag(Bt + (size_t)n_base * H_, H_, k, lane);
    v16bf b1 = load_b_frag(Bt + (size_t)(n_base + 16) * H_, H_, k, lane);
    acc[0][0] = wmma_bf16(a0, b0, acc[0][0]);
    acc[0][1] = wmma_bf16(a0, b1, acc[0][1]);
    acc[1][0] = wmma_bf16(a1, b0, acc[1][0]);
    acc[1][1] = wmma_bf16(a1, b1, acc[1][1]);
  }
  int half = lane >> 4, nn = lane & 15;
#pragma unroll
  for (int i = 0; i < 2; ++i)
#pragma unroll
    for (int j = 0; j < 2; ++j) {
      int col = n_base + 16 * j + nn;
      float bv = bias[col];
#pragma unroll
      for (int r = 0; r < 8; ++r) {
        int row = m_base + 16 * i + r + 8 * half;
        float v = (acc[i][j][r] + bv) * scale;
        if (OutVt) {                 // store as V^T : [B][H][S]
          int b = row >> 11, s = row & (S_ - 1);
          OutVt[((size_t)(b * H_ + col) << 11) + s] = to_bf16(v);
        } else {
          OutRM[(size_t)row * H_ + col] = to_bf16(v);
        }
      }
    }
}

// ---------------- Kernel 3: flash attention (16 query rows / block) -------------------
__global__ void __launch_bounds__(256)
attn_kernel(const bf16* __restrict__ Q, const bf16* __restrict__ K,
            const bf16* __restrict__ Vt, const int* __restrict__ mask,
            bf16* __restrict__ Ctx) {
  __shared__ bf16  lds_q[16 * H_];
  __shared__ float lds_s[16 * 128];
  __shared__ bf16  lds_p[16 * 128];
  __shared__ float lds_m[16], lds_l[16], lds_scale[16];

  int b = blockIdx.y;
  int qbase = blockIdx.x * 16;
  int tid = threadIdx.x;
  int lane = tid & 31, w = tid >> 5;
  int half = lane >> 4, nn = lane & 15;
  int hbase = w * 96;

  // ---- async-tensor staging of the Q tile (24 KB), ASYNCcnt-tracked ----
  {
    const char* Qrow = (const char*)(Q + (size_t)(b * S_ + qbase) * H_);
    uint32_t lds_base = (uint32_t)(uintptr_t)(&lds_q[0]);
    for (int i = tid; i < (16 * H_ * 2) / 16; i += 256)
      async_copy_b128(lds_base + i * 16, Qrow + (size_t)i * 16);
    wait_async0();
  }
  if (tid < 16) { lds_m[tid] = -1e38f; lds_l[tid] = 0.0f; }
  __syncthreads();

  v8f ctx[6] = {};

  for (int kb = 0; kb < S_; kb += 128) {
    int keycol = kb + w * 16 + nn;
    int mk = mask[b * S_ + keycol];

    // --- scores: 16x16 tile per wave, K-reduce over H ---
    v8f sfr = {};
    const bf16* Kp = K + (size_t)(b * S_ + kb + w * 16) * H_;
    for (int k = 0; k < H_; k += 32) {
      v16bf a = load_a_frag(lds_q, H_, k, lane);
      v16bf bb = load_b_frag(Kp, H_, k, lane);
      sfr = wmma_bf16(a, bb, sfr);
    }
#pragma unroll
    for (int r = 0; r < 8; ++r) {
      float sv = mk ? sfr[r] : -1e38f;
      sfr[r] = sv;
      lds_s[(r + 8 * half) * 128 + w * 16 + nn] = sv;
    }
    __syncthreads();                                  // scores visible

    if (w == 0) {                                     // row stats (2 lanes/row)
      int row = lane & 15;
      int jo = (lane >> 4) * 64;
      float bm = -1e38f;
      for (int j = 0; j < 64; ++j) bm = fmaxf(bm, lds_s[row * 128 + jo + j]);
      bm = fmaxf(bm, __shfl_xor(bm, 16, 32));
      if (lane < 16) {
        float mo = lds_m[row];
        float mn = fmaxf(mo, bm);
        lds_m[row] = mn;
        lds_scale[row] = __expf(mo - mn);
      }
    }
    __syncthreads();                                  // stats visible

#pragma unroll
    for (int r = 0; r < 8; ++r) {                     // probs (masked -> 0)
      int row = r + 8 * half;
      float p = mk ? __expf(sfr[r] - lds_m[row]) : 0.0f;
      lds_p[row * 128 + w * 16 + nn] = to_bf16(p);
    }
    __syncthreads();                                  // probs visible

    if (w == 1) {                                     // running denominator
      int row = lane & 15;
      int jo = (lane >> 4) * 64;
      float s = 0.0f;
      for (int j = 0; j < 64; ++j) s += from_bf16(lds_p[row * 128 + jo + j]);
      s += __shfl_xor(s, 16, 32);
      if (lane < 16) lds_l[row] = lds_l[row] * lds_scale[row] + s;
    }

#pragma unroll
    for (int r = 0; r < 8; ++r) {                     // rescale old ctx
      float f = lds_scale[r + 8 * half];
#pragma unroll
      for (int fr = 0; fr < 6; ++fr) ctx[fr][r] *= f;
    }

    for (int kk = 0; kk < 128; kk += 32) {            // ctx += P * V
      v16bf a = load_a_frag(lds_p, 128, kk, lane);
#pragma unroll
      for (int f = 0; f < 6; ++f) {
        v16bf bb = load_b_frag(Vt + (size_t)(b * H_ + hbase + 16 * f) * S_ + kb, S_, kk, lane);
        ctx[f] = wmma_bf16(a, bb, ctx[f]);
      }
    }
  }
  __syncthreads();                                    // final l visible

#pragma unroll
  for (int r = 0; r < 8; ++r) {
    int row = r + 8 * half;
    float li = lds_l[row];
    float inv = li > 0.0f ? 1.0f / li : 0.0f;
#pragma unroll
    for (int f = 0; f < 6; ++f) {
      float v = ctx[f][r] * inv;
      Ctx[(size_t)(b * S_ + qbase + row) * H_ + hbase + 16 * f + nn] = to_bf16(v);
    }
  }
}

// ---------------- Kernel 4a: output GEMM + bias + residual (fp32 out) -----------------
__global__ void __launch_bounds__(256)
out_gemm(const bf16* __restrict__ A, const bf16* __restrict__ Bt,
         const float* __restrict__ bias, const float* __restrict__ resid,
         float* __restrict__ Hout) {
  int lane = threadIdx.x & 31;
  int w = threadIdx.x >> 5;
  int m_base = blockIdx.x * 128 + (w & 3) * 32;
  int n_base = blockIdx.y * 64 + (w >> 2) * 32;

  v8f acc[2][2] = {};
  for (int k = 0; k < H_; k += 32) {
    v16bf a0 = load_a_frag(A + (size_t)m_base * H_, H_, k, lane);
    v16bf a1 = load_a_frag(A + (size_t)(m_base + 16) * H_, H_, k, lane);
    v16bf b0 = load_b_frag(Bt + (size_t)n_base * H_, H_, k, lane);
    v16bf b1 = load_b_frag(Bt + (size_t)(n_base + 16) * H_, H_, k, lane);
    acc[0][0] = wmma_bf16(a0, b0, acc[0][0]);
    acc[0][1] = wmma_bf16(a0, b1, acc[0][1]);
    acc[1][0] = wmma_bf16(a1, b0, acc[1][0]);
    acc[1][1] = wmma_bf16(a1, b1, acc[1][1]);
  }
  int half = lane >> 4, nn = lane & 15;
#pragma unroll
  for (int i = 0; i < 2; ++i)
#pragma unroll
    for (int j = 0; j < 2; ++j) {
      int col = n_base + 16 * j + nn;
      float bv = bias[col];
#pragma unroll
      for (int r = 0; r < 8; ++r) {
        int row = m_base + 16 * i + r + 8 * half;
        size_t idx = (size_t)row * H_ + col;
        Hout[idx] = acc[i][j][r] + bv + resid[idx];
      }
    }
}

// ---------------- Kernel 4b: LayerNorm over H=768 (one block per row) ----------------
__global__ void __launch_bounds__(256)
ln_kernel(const float* __restrict__ Hb, const float* __restrict__ gamma,
          const float* __restrict__ beta, float* __restrict__ Out) {
  __shared__ float red[256], red2[256];
  int row = blockIdx.x, tid = threadIdx.x;
  const float* hp = Hb + (size_t)row * H_;
  float x0 = hp[tid], x1 = hp[tid + 256], x2 = hp[tid + 512];
  red[tid] = x0 + x1 + x2;
  red2[tid] = x0 * x0 + x1 * x1 + x2 * x2;
  __syncthreads();
  for (int off = 128; off > 0; off >>= 1) {
    if (tid < off) { red[tid] += red[tid + off]; red2[tid] += red2[tid + off]; }
    __syncthreads();
  }
  float mu = red[0] * (1.0f / H_);
  float var = red2[0] * (1.0f / H_) - mu * mu;
  float rs = rsqrtf(var + LN_EPS);
  float* op = Out + (size_t)row * H_;
  op[tid]       = (x0 - mu) * rs * gamma[tid]       + beta[tid];
  op[tid + 256] = (x1 - mu) * rs * gamma[tid + 256] + beta[tid + 256];
  op[tid + 512] = (x2 - mu) * rs * gamma[tid + 512] + beta[tid + 512];
}

extern "C" void kernel_launch(void* const* d_in, const int* in_sizes, int n_in,
                              void* d_out, int out_size, void* d_ws, size_t ws_size,
                              hipStream_t stream) {
  const float* X     = (const float*)d_in[0];
  const int*   mask  = (const int*)d_in[1];
  const float* Wq    = (const float*)d_in[2];
  const float* bq    = (const float*)d_in[3];
  const float* Wk    = (const float*)d_in[4];
  const float* bk    = (const float*)d_in[5];
  const float* Wv    = (const float*)d_in[6];
  const float* bv    = (const float*)d_in[7];
  const float* Wo    = (const float*)d_in[8];
  const float* bo    = (const float*)d_in[9];
  const float* gamma = (const float*)d_in[10];
  const float* beta  = (const float*)d_in[11];
  float* out = (float*)d_out;

  const size_t NBSH = (size_t)B_ * S_ * H_;   // 12,582,912
  const size_t HH   = (size_t)H_ * H_;        // 589,824
  char* ws = (char*)d_ws;
  bf16* Xb    = (bf16*)ws;                                   // bf16 X
  bf16* WtAll = (bf16*)(ws + NBSH * 2);                      // 4 transposed weights
  bf16* Qb    = (bf16*)(ws + NBSH * 2 + 4 * HH * 2);
  bf16* Kb    = Qb + NBSH;
  bf16* Vt    = Kb + NBSH;                                   // [B][H][S]
  bf16* Ctx   = Xb;                                          // alias (X bf16 dead after QKV)
  float* Hbuf = (float*)Qb;                                  // alias (Q+K dead after attn)

  cvt_kernel<<<4096, 256, 0, stream>>>(X, Xb, (int)NBSH);
  dim3 tg(H_ / 16, H_ / 16);
  wtrans_kernel<<<tg, dim3(16, 16), 0, stream>>>(Wq, WtAll + 0 * HH);
  wtrans_kernel<<<tg, dim3(16, 16), 0, stream>>>(Wk, WtAll + 1 * HH);
  wtrans_kernel<<<tg, dim3(16, 16), 0, stream>>>(Wv, WtAll + 2 * HH);
  wtrans_kernel<<<tg, dim3(16, 16), 0, stream>>>(Wo, WtAll + 3 * HH);

  const float qscale = 0.03608439182435161f;  // 1/sqrt(768), folded into Q
  dim3 gg(B_ * S_ / 128, H_ / 64);
  gemm_bf16<<<gg, 256, 0, stream>>>(Xb, WtAll + 0 * HH, bq, Qb, nullptr, qscale);
  gemm_bf16<<<gg, 256, 0, stream>>>(Xb, WtAll + 1 * HH, bk, Kb, nullptr, 1.0f);
  gemm_bf16<<<gg, 256, 0, stream>>>(Xb, WtAll + 2 * HH, bv, nullptr, Vt, 1.0f);

  attn_kernel<<<dim3(S_ / 16, B_), 256, 0, stream>>>(Qb, Kb, Vt, mask, Ctx);

  out_gemm<<<gg, 256, 0, stream>>>(Ctx, WtAll + 3 * HH, bo, X, Hbuf);
  ln_kernel<<<B_ * S_, 256, 0, stream>>>(Hbuf, gamma, beta, out);
}
